// SUCode_36412732735533
// MI455X (gfx1250) — matrix operand order repaired and checked by reading, not compiled, not run on hardware
//
#include <hip/hip_runtime.h>

typedef __attribute__((ext_vector_type(16))) _Float16 v16h;
typedef __attribute__((ext_vector_type(8)))  _Float16 v8h;
typedef __attribute__((ext_vector_type(8)))  float    v8f;

#define B_   8
#define C_   256
#define H_   64
#define W_   64
#define K_   4
#define NE_  1024
#define HW_  (H_*W_)          // 4096
#define N_   (B_*HW_)         // 32768

// ---------------------------------------------------------------------------
// Kernel 1: z [B,C,H,W] f32  ->  zh [N, C] f16   (tiled LDS transpose)
// grid: B * (C/32) * (HW/32) = 8*8*128 = 8192 blocks, block (32,8)
// ---------------------------------------------------------------------------
__global__ void k_transpose(const float* __restrict__ z, _Float16* __restrict__ zh) {
    __shared__ _Float16 tile[32][33];
    int blk = blockIdx.x;
    int hwT = blk % 128; blk /= 128;
    int cT  = blk % 8;   int b = blk / 8;
    int tx = threadIdx.x, ty = threadIdx.y;
    const float* src = z + ((size_t)b * C_ + cT * 32) * HW_ + hwT * 32;
#pragma unroll
    for (int i = 0; i < 32; i += 8)
        tile[ty + i][tx] = (_Float16)src[(size_t)(ty + i) * HW_ + tx];
    __syncthreads();
    _Float16* dst = zh + ((size_t)b * HW_ + hwT * 32) * C_ + cT * 32;
#pragma unroll
    for (int i = 0; i < 32; i += 8)
        dst[(size_t)(ty + i) * C_ + tx] = tile[tx][ty + i];
}

// ---------------------------------------------------------------------------
// Kernel 2: codebooks f32 -> f16 copy + per-code squared norm
// grid: K*NE = 4096 blocks, 256 threads (one d per thread)
// ---------------------------------------------------------------------------
__global__ void k_cbprep(const float* __restrict__ cb, _Float16* __restrict__ cbh,
                         float* __restrict__ cbnorm) {
    int row = blockIdx.x;                 // k*NE_ + e
    int d   = threadIdx.x;
    float v = cb[(size_t)row * C_ + d];
    cbh[(size_t)row * C_ + d] = (_Float16)v;
    __shared__ float red[256];
    red[d] = v * v;
    __syncthreads();
    for (int s = 128; s > 0; s >>= 1) { if (d < s) red[d] += red[d + s]; __syncthreads(); }
    if (d == 0) cbnorm[row] = red[0];
}

__device__ inline v16h cat8(v8h lo, v8h hi) {
    v16h r;
#pragma unroll
    for (int j = 0; j < 8; ++j) { r[j] = lo[j]; r[j + 8] = hi[j]; }
    return r;
}

// ---------------------------------------------------------------------------
// Kernel 3: WMMA distance + argmin.
// One wave = one (codebook k, 32-token tile): two M-tiles share every B
// fragment -> 16 WMMAs per e-tile on 8 streamed B loads (2x arithmetic
// intensity vs 16-token tiles). B is streamed one fragment at a time to keep
// the A fragments (128 VGPRs) resident -- no spills.
// Jobs = 4 * 1024 = 4096.  block = 128 threads (4 waves), grid = 1024.
// dist[m,e] = ||e||^2 - 2 * sum_d zh[m,d]*cbh[e,d]   (||x||^2 dropped: const/row)
// ---------------------------------------------------------------------------
__global__ void __launch_bounds__(128, 1)
k_argmin_wmma(const _Float16* __restrict__ zh,
              const _Float16* __restrict__ cbh,
              const float* __restrict__ cbnorm,
              int* __restrict__ idxbuf) {
    int wid  = threadIdx.x >> 5;
    int job  = blockIdx.x * 4 + wid;          // 0..4095
    int k    = job >> 10;                     // 0..3
    int tile = job & 1023;                    // 0..1023
    int n0   = tile * 32;

    int lane = threadIdx.x & 31;
    int half = lane >> 4;                     // 0 or 1
    int lm   = lane & 15;

    // A fragments: row M = n0+lm (tile 0) / n0+16+lm (tile 1).
    // lane<16: K kc+0..7 & kc+16..23, lane>=16: kc+8..15 & kc+24..31
    const _Float16* arow0 = zh + (size_t)(n0 + lm) * C_;
    const _Float16* arow1 = arow0 + (size_t)16 * C_;
    v16h a0[8], a1[8];
#pragma unroll
    for (int c = 0; c < 8; ++c) {
        int kc = c * 32;
        a0[c] = cat8(*(const v8h*)(arow0 + kc + half * 8),
                     *(const v8h*)(arow0 + kc + 16 + half * 8));
        a1[c] = cat8(*(const v8h*)(arow1 + kc + half * 8),
                     *(const v8h*)(arow1 + kc + 16 + half * 8));
    }

    float bestv0[8], bestv1[8];
    int   besti0[8], besti1[8];
#pragma unroll
    for (int r = 0; r < 8; ++r) {
        bestv0[r] = 3.4e38f; besti0[r] = 0;
        bestv1[r] = 3.4e38f; besti1[r] = 0;
    }

    const _Float16* cbk = cbh + (size_t)k * NE_ * C_;
    const float*    cnk = cbnorm + k * NE_;

#pragma unroll 1
    for (int et = 0; et < 64; ++et) {
        int e0 = et * 16;
        // B fragment: col N = e0+lm; lane<16 holds K kc+0..7 & kc+16..23, lane>=16 kc+8..15 & kc+24..31
        const _Float16* brow = cbk + (size_t)(e0 + lm) * C_ + half * 8;
        float s = cnk[e0 + lm];

        v8f acc0 = {}, acc1 = {};
#pragma unroll
        for (int c = 0; c < 8; ++c) {
            // Stream one B fragment; it feeds both M-tiles' WMMAs.
            v16h bv = cat8(*(const v8h*)(brow + c * 32), *(const v8h*)(brow + c * 32 + 16));
            acc0 = __builtin_amdgcn_wmma_f32_16x16x32_f16(
                false, a0[c], false, bv, (short)0, acc0, false, false);
            acc1 = __builtin_amdgcn_wmma_f32_16x16x32_f16(
                false, a1[c], false, bv, (short)0, acc1, false, false);
        }

        int ei = e0 + lm;
#pragma unroll
        for (int r = 0; r < 8; ++r) {
            float d0 = s - 2.0f * acc0[r];
            float d1 = s - 2.0f * acc1[r];
            if (d0 < bestv0[r]) { bestv0[r] = d0; besti0[r] = ei; }
            if (d1 < bestv1[r]) { bestv1[r] = d1; besti1[r] = ei; }
        }
    }

    // Cross-lane argmin across the 16 columns (within each 16-lane half).
    // C/D layout: element (M = r + 8*half, N = lm). Tiebreak: smaller index (jnp.argmin).
#pragma unroll
    for (int r = 0; r < 8; ++r) {
        float v0 = bestv0[r]; int i0 = besti0[r];
        float v1 = bestv1[r]; int i1 = besti1[r];
#pragma unroll
        for (int m = 1; m < 16; m <<= 1) {
            float ov0 = __shfl_xor(v0, m, 32);
            int   oi0 = __shfl_xor(i0, m, 32);
            float ov1 = __shfl_xor(v1, m, 32);
            int   oi1 = __shfl_xor(i1, m, 32);
            if (ov0 < v0 || (ov0 == v0 && oi0 < i0)) { v0 = ov0; i0 = oi0; }
            if (ov1 < v1 || (ov1 == v1 && oi1 < i1)) { v1 = ov1; i1 = oi1; }
        }
        if (lm == 0) {
            int m = r + 8 * half;
            idxbuf[(size_t)k * N_ + n0 + m]      = i0;
            idxbuf[(size_t)k * N_ + n0 + 16 + m] = i1;
        }
    }
}

// ---------------------------------------------------------------------------
// Kernel 4: fused output + loss partials.
// grid = 512 blocks (one (b,h) row = 64 tokens), 256 threads = 64 w x 4 d-groups
// ---------------------------------------------------------------------------
__global__ void k_fuse(const float* __restrict__ z, const float* __restrict__ wl,
                       const float* __restrict__ cb, const int* __restrict__ idxbuf,
                       float* __restrict__ out, float* __restrict__ partial) {
    int grp = blockIdx.x;     // 0..511
    int b   = grp >> 6;
    int h   = grp & 63;
    int tid = threadIdx.x;
    int w   = tid & 63;
    int dg  = tid >> 6;       // 0..3
    int n   = (b * 64 + h) * 64 + w;

    // softmax over K=4
    float lg[4];
#pragma unroll
    for (int k = 0; k < 4; ++k)
        lg[k] = wl[(((size_t)b * K_ + k) * H_ + h) * W_ + w];
    float mx = fmaxf(fmaxf(lg[0], lg[1]), fmaxf(lg[2], lg[3]));
    float ex[4], se = 0.f;
#pragma unroll
    for (int k = 0; k < 4; ++k) { ex[k] = __expf(lg[k] - mx); se += ex[k]; }
    float inv = 1.0f / se;
#pragma unroll
    for (int k = 0; k < 4; ++k) ex[k] *= inv;

    const float* cbp[4];
#pragma unroll
    for (int k = 0; k < 4; ++k) {
        int id = idxbuf[(size_t)k * N_ + n];
        cbp[k] = cb + ((size_t)k * NE_ + id) * C_;
    }

    float lsum = 0.f;
    for (int d = dg; d < C_; d += 4) {
        size_t zi = (((size_t)b * C_ + d) * H_ + h) * W_ + w;
        float zv = z[zi];
        float f = 0.f;
#pragma unroll
        for (int k = 0; k < 4; ++k) {
            float q = cbp[k][d];
            f += ex[k] * q;
            float df = q - zv;
            lsum += df * df;
        }
        out[zi] = f;
    }

    __shared__ float red[256];
    red[tid] = lsum;
    __syncthreads();
    for (int s = 128; s > 0; s >>= 1) { if (tid < s) red[tid] += red[tid + s]; __syncthreads(); }
    if (tid == 0) partial[grp] = red[0];
}

// ---------------------------------------------------------------------------
// Kernel 5: reduce 512 partials -> scalar loss = 1.25/(N*D) * sum
// ---------------------------------------------------------------------------
__global__ void k_loss(const float* __restrict__ partial, float* __restrict__ outLoss) {
    __shared__ float red[256];
    int t = threadIdx.x;
    red[t] = partial[t] + partial[t + 256];
    __syncthreads();
    for (int s = 128; s > 0; s >>= 1) { if (t < s) red[t] += red[t + s]; __syncthreads(); }
    if (t == 0) *outLoss = red[0] * (1.25f / ((float)N_ * (float)C_));
}

// ---------------------------------------------------------------------------
extern "C" void kernel_launch(void* const* d_in, const int* in_sizes, int n_in,
                              void* d_out, int out_size, void* d_ws, size_t ws_size,
                              hipStream_t stream) {
    const float* z  = (const float*)d_in[0];   // [8,256,64,64]
    const float* wl = (const float*)d_in[1];   // [8,4,64,64]
    const float* cb = (const float*)d_in[2];   // [4,1024,256]
    float* out = (float*)d_out;                // fused (8388608) + loss (1)

    // workspace layout (~18.6 MB)
    char* ws = (char*)d_ws;
    _Float16* zh     = (_Float16*)(ws);                 // 16 MB
    _Float16* cbh    = (_Float16*)(ws + 16777216);      //  2 MB
    float*    cbnorm = (float*)   (ws + 18874368);      // 16 KB
    int*      idxbuf = (int*)     (ws + 18890752);      // 512 KB
    float*    partial= (float*)   (ws + 19415040);      //  2 KB

    hipLaunchKernelGGL(k_transpose,   dim3(8192), dim3(32, 8), 0, stream, z, zh);
    hipLaunchKernelGGL(k_cbprep,      dim3(4096), dim3(256),   0, stream, cb, cbh, cbnorm);
    hipLaunchKernelGGL(k_argmin_wmma, dim3(1024), dim3(128),   0, stream, zh, cbh, cbnorm, idxbuf);
    hipLaunchKernelGGL(k_fuse,        dim3(512),  dim3(256),   0, stream, z, wl, cb, idxbuf, out, partial);
    hipLaunchKernelGGL(k_loss,        dim3(1),    dim3(256),   0, stream, partial, out + (size_t)N_ * C_);
}
